// MolecularGNN_52621939311117
// MI455X (gfx1250) — matrix operand order repaired and checked by reading, not compiled, not run on hardware
//
#include <hip/hip_runtime.h>
#include <hip/hip_bf16.h>

#define NNODES 50000
#define NEDGES 800000
#define ENTOT  850000          // E + N (with self loops)
#define HD_    256
#define NHEAD  4
#define NLAY   3
#define NGRAPH 512
#define NEGS   0.2f
#define INV_STD 0.9999950000374997f   // 1/sqrt(1+1e-5)

typedef __attribute__((ext_vector_type(16))) __bf16 v16bf;
typedef __attribute__((ext_vector_type(8)))  float  v8f;
typedef unsigned int v4u  __attribute__((ext_vector_type(4)));
typedef int          v4i  __attribute__((ext_vector_type(4)));
typedef int          v8i  __attribute__((ext_vector_type(8)));

#ifndef __has_builtin
#define __has_builtin(x) 0
#endif
#if __has_builtin(__builtin_amdgcn_tensor_load_to_lds)
#define HAVE_TDM 1
#else
#define HAVE_TDM 0
#endif

static __device__ __forceinline__ unsigned short f2bf(float f) {
  unsigned u = __float_as_uint(f);
  u += 0x7fffu + ((u >> 16) & 1u);           // round-to-nearest-even
  return (unsigned short)(u >> 16);
}

static __device__ __forceinline__ void atomicMaxF(float* addr, float val) {
  unsigned int* ua = (unsigned int*)addr;
  unsigned int old = __float_as_uint(*addr);
  while (__uint_as_float(old) < val) {
    unsigned int assumed = old;
    old = atomicCAS(ua, assumed, __float_as_uint(val));
    if (old == assumed) break;
  }
}

// ---------------------------------------------------------------- utilities
__global__ void k_fill(float* p, float v, long n) {
  long i = (long)blockIdx.x * blockDim.x + threadIdx.x;
  if (i < n) p[i] = v;
}
__global__ void k_f2bf(const float* __restrict__ s, unsigned short* __restrict__ d, long n) {
  long i = (long)blockIdx.x * blockDim.x + threadIdx.x;
  if (i < n) d[i] = f2bf(s[i]);
}

// ---------------------------------------------------------------- embeddings
__global__ void k_node_embed(const float* __restrict__ x, const float* __restrict__ Wn,
                             const float* __restrict__ bnb, float* __restrict__ h) {
  long i = (long)blockIdx.x * blockDim.x + threadIdx.x;
  if (i >= (long)NNODES * HD_) return;
  int n = (int)(i >> 8), ch = (int)(i & 255);
  float s = bnb[ch];
  #pragma unroll
  for (int k = 0; k < 9; ++k) s += x[n * 9 + k] * Wn[k * HD_ + ch];
  h[i] = s;
}

__global__ void k_deg(const int* __restrict__ ei, float* __restrict__ deg) {
  int e = blockIdx.x * blockDim.x + threadIdx.x;
  if (e < NEDGES) atomicAdd(&deg[ei[NEDGES + e]], 1.0f);
}

__global__ void k_edge_embed(const float* __restrict__ eattr, const float* __restrict__ We,
                             const float* __restrict__ beb, const int* __restrict__ ei,
                             unsigned short* __restrict__ eabf, float* __restrict__ loop_sum) {
  long i = (long)blockIdx.x * blockDim.x + threadIdx.x;
  if (i >= (long)NEDGES * HD_) return;
  int e = (int)(i >> 8), ch = (int)(i & 255);
  float s = beb[ch];
  #pragma unroll
  for (int k = 0; k < 5; ++k) s += eattr[e * 5 + k] * We[k * HD_ + ch];
  eabf[i] = f2bf(s);
  atomicAdd(&loop_sum[(long)ei[NEDGES + e] * HD_ + ch], s);   // for self-loop mean attr
}

__global__ void k_loop_attr(const float* __restrict__ loop_sum, const float* __restrict__ deg,
                            unsigned short* __restrict__ eabf) {
  long i = (long)blockIdx.x * blockDim.x + threadIdx.x;
  if (i >= (long)NNODES * HD_) return;
  int n = (int)(i >> 8);
  float v = loop_sum[i] / fmaxf(deg[n], 1.0f);
  eabf[(long)(NEDGES + n) * HD_ + (i & 255)] = f2bf(v);
}

// fold lin_edge_w (256x256) with att_edge (4x64) -> [256,16] bf16 (cols 4..15 zero)
__global__ void k_watt_e(const float* __restrict__ lew, const float* __restrict__ ate,
                         unsigned short* __restrict__ wout) {
  int i = blockIdx.x * blockDim.x + threadIdx.x;
  if (i >= 256 * 16) return;
  int k = i >> 4, col = i & 15;
  float s = 0.0f;
  if (col < NHEAD) {
    #pragma unroll 16
    for (int c = 0; c < 64; ++c) s += lew[k * HD_ + col * 64 + c] * ate[col * 64 + c];
  }
  wout[k * 16 + col] = f2bf(s);
}

// ---------------------------------------------------------------- WMMA GEMM
// C[M,NB](f32) = A[M,256](bf16) @ B[256,NB](bf16).  One 16x16 tile per wave,
// K=256 in 8 steps of v_wmma_f32_16x16x32_bf16.  B strip (256x16 bf16) is pulled
// into LDS by the Tensor Data Mover (one wave issues tensor_load_to_lds, waits
// s_wait_tensorcnt), then transposed in LDS to the padded fragment layout.
// Requires M % 16 == 0 (both call sites satisfy this).
#define LDSK 264   // 256 halves + 8 pad (16B aligned, staggers banks)
template <int NB>
__global__ void __launch_bounds__(256)
k_gemm_bf16(const unsigned short* __restrict__ A, const unsigned short* __restrict__ B,
            float* __restrict__ C, int M) {
  __shared__ unsigned short braw[256 * 16];   // TDM destination, [k][n]
  __shared__ unsigned short bsh[16 * LDSK];   // transposed, [n][k]
  const int nbase = blockIdx.x * 16;

#if HAVE_TDM
  if (threadIdx.x < 32) {                     // wave 0 drives the TDM
    unsigned long long ga = (unsigned long long)(const void*)(B + nbase);
    unsigned int lds = (unsigned int)(unsigned long long)(void*)&braw[0];
    v4u g0;
    g0[0] = 1u;                                          // count=1, no gather
    g0[1] = lds;                                         // lds_addr
    g0[2] = (unsigned int)(ga & 0xffffffffu);            // global_addr[31:0]
    g0[3] = (unsigned int)((ga >> 32) & 0x1ffffffu)      // global_addr[56:32]
            | 0x80000000u;                               // type=2 ("image")
    v8i g1;
    g1[0] = 0x00010000;                                  // data_size=1 (2B), wg_mask=0
    g1[1] = (int)(((unsigned)NB & 0xffffu) << 16);       // tensor_dim0 lo16
    g1[2] = (int)((((unsigned)NB >> 16) & 0xffffu)       // tensor_dim0 hi16
            | (256u << 16));                             // tensor_dim1 lo16 (=256 rows)
    g1[3] = (int)(((256u >> 16) & 0xffffu)               // tensor_dim1 hi16
            | (16u << 16));                              // tile_dim0 = 16 cols
    g1[4] = (int)256u;                                   // tile_dim1 = 256 rows
    g1[5] = (int)NB;                                     // tensor_dim0_stride lo32
    g1[6] = 0;
    g1[7] = 0;
    v4i gz = {0, 0, 0, 0};
#if defined(__clang_major__) && (__clang_major__ >= 23)
    v8i gz8 = {0, 0, 0, 0, 0, 0, 0, 0};
    __builtin_amdgcn_tensor_load_to_lds(g0, g1, gz, gz, gz8, 0);
#else
    __builtin_amdgcn_tensor_load_to_lds(g0, g1, gz, gz, 0);
#endif
    __builtin_amdgcn_s_wait_tensorcnt(0);
  }
  __syncthreads();
  {   // LDS transpose [k][n] -> [n][k]; thread t handles row k=t
    int k = threadIdx.x;
    #pragma unroll
    for (int n = 0; n < 16; ++n) bsh[n * LDSK + k] = braw[k * 16 + n];
  }
#else
  {   // fallback: cooperative transpose-stage straight from global
    int k = threadIdx.x;
    const unsigned short* src = B + k * NB + nbase;
    #pragma unroll
    for (int n = 0; n < 16; ++n) bsh[n * LDSK + k] = src[n];
  }
#endif
  __syncthreads();

  const int wave = threadIdx.x >> 5;
  const int lane = threadIdx.x & 31;
  const int nrow = lane & 15;
  const int hi   = lane >> 4;           // 0 or 1
  union Frag { uint4 q[2]; v16bf v; };

  // B fragments for all 8 k-steps (kept in VGPRs, reused per tile)
  Frag bf[8];
  #pragma unroll
  for (int kk = 0; kk < 8; ++kk) {
    const unsigned short* bp = &bsh[nrow * LDSK + kk * 32 + hi * 16];
    bf[kk].q[0] = *(const uint4*)(bp);
    bf[kk].q[1] = *(const uint4*)(bp + 8);
  }

  long mtile = (long)blockIdx.y * 8 + wave;
  long mbase = mtile * 16;
  if (mbase >= M) return;               // wave-uniform exit (M % 16 == 0)
  const unsigned short* ap = A + (mbase + nrow) * 256 + hi * 16;
  __builtin_prefetch(ap, 0, 0);         // global_prefetch_b8

  v8f acc = {0.f, 0.f, 0.f, 0.f, 0.f, 0.f, 0.f, 0.f};
  #pragma unroll
  for (int kk = 0; kk < 8; ++kk) {
    Frag af;
    af.q[0] = *(const uint4*)(ap + kk * 32);
    af.q[1] = *(const uint4*)(ap + kk * 32 + 8);
    acc = __builtin_amdgcn_wmma_f32_16x16x32_bf16(
        false, af.v, false, bf[kk].v, (short)0, acc, false, false);
  }
  float* cp = C + (mbase + (hi << 3)) * NB + nbase + nrow;
  #pragma unroll
  for (int j = 0; j < 8; ++j)           // C/D layout: vgpr j, lanes0-15 M=j, lanes16-31 M=j+8
    cp[(long)j * NB] = acc[j];
}

// ---------------------------------------------------------------- attention
__global__ void k_att_nd(const float* __restrict__ xh, const float* __restrict__ as_,
                         const float* __restrict__ ad_, float* __restrict__ a_src,
                         float* __restrict__ a_dst) {
  int i = blockIdx.x * blockDim.x + threadIdx.x;
  if (i >= NNODES * NHEAD) return;
  int n = i >> 2, hd = i & 3;
  const float* xr = xh + (long)n * HD_ + hd * 64;
  float ss = 0.f, sd = 0.f;
  #pragma unroll 16
  for (int c = 0; c < 64; ++c) { ss += xr[c] * as_[hd * 64 + c]; sd += xr[c] * ad_[hd * 64 + c]; }
  a_src[i] = ss; a_dst[i] = sd;
}

static __device__ __forceinline__ void edge_rc(const int* ei, int e, int& r, int& c) {
  if (e < NEDGES) { r = ei[e]; c = ei[NEDGES + e]; }
  else            { r = c = e - NEDGES; }
}

__global__ void k_alpha(const int* __restrict__ ei, const float* __restrict__ a_src,
                        const float* __restrict__ a_dst, const float* __restrict__ aeC,
                        float* __restrict__ alpha, float* __restrict__ seg_m) {
  long i = (long)blockIdx.x * blockDim.x + threadIdx.x;
  if (i >= (long)ENTOT * NHEAD) return;
  int e = (int)(i >> 2), hd = (int)(i & 3), r, c;
  edge_rc(ei, e, r, c);
  float a = a_src[r * 4 + hd] + a_dst[c * 4 + hd] + aeC[(long)e * 16 + hd];
  a = (a >= 0.f) ? a : NEGS * a;
  alpha[i] = a;
  atomicMaxF(&seg_m[c * 4 + hd], a);
}

__global__ void k_exp(const int* __restrict__ ei, float* __restrict__ alpha,
                      const float* __restrict__ seg_m, float* __restrict__ seg_s) {
  long i = (long)blockIdx.x * blockDim.x + threadIdx.x;
  if (i >= (long)ENTOT * NHEAD) return;
  int e = (int)(i >> 2), hd = (int)(i & 3), r, c;
  edge_rc(ei, e, r, c);
  float ex = __expf(alpha[i] - seg_m[c * 4 + hd]);
  alpha[i] = ex;
  atomicAdd(&seg_s[c * 4 + hd], ex);
}

__global__ void k_wnorm(const int* __restrict__ ei, float* __restrict__ alpha,
                        const float* __restrict__ seg_s) {
  long i = (long)blockIdx.x * blockDim.x + threadIdx.x;
  if (i >= (long)ENTOT * NHEAD) return;
  int e = (int)(i >> 2), hd = (int)(i & 3), r, c;
  edge_rc(ei, e, r, c);
  alpha[i] = alpha[i] / (seg_s[c * 4 + hd] + 1e-16f);
}

// gather xh[r], weight by alpha, scatter-add into agg[c]  (xh & agg are L2-resident)
__global__ void k_msg(const int* __restrict__ ei, const float* __restrict__ xh,
                      const float* __restrict__ alpha, float* __restrict__ agg) {
  long i = (long)blockIdx.x * blockDim.x + threadIdx.x;
  if (i >= (long)ENTOT * HD_) return;
  int e = (int)(i >> 8), ch = (int)(i & 255), r, c;
  edge_rc(ei, e, r, c);
  float w = alpha[e * 4 + (ch >> 6)];
  atomicAdd(&agg[(long)c * HD_ + ch], xh[(long)r * HD_ + ch] * w);
}

__global__ void k_bn_relu(const float* __restrict__ agg, const float* __restrict__ bias,
                          const float* __restrict__ gamma, const float* __restrict__ beta,
                          float* __restrict__ hn) {
  long i = (long)blockIdx.x * blockDim.x + threadIdx.x;
  if (i >= (long)NNODES * HD_) return;
  int ch = (int)(i & 255);
  float v = (agg[i] + bias[ch]) * INV_STD * gamma[ch] + beta[ch];
  hn[i] = fmaxf(v, 0.f);
}

// ---------------------------------------------------------------- pooling
__global__ void k_cnt(const int* __restrict__ batch, float* __restrict__ cnt) {
  int n = blockIdx.x * blockDim.x + threadIdx.x;
  if (n < NNODES) atomicAdd(&cnt[batch[n]], 1.0f);
}
__global__ void k_pool(const int* __restrict__ batch, const float* __restrict__ h,
                       float* __restrict__ outp) {
  long i = (long)blockIdx.x * blockDim.x + threadIdx.x;
  if (i >= (long)NNODES * HD_) return;
  atomicAdd(&outp[(long)batch[i >> 8] * HD_ + (i & 255)], h[i]);
}
__global__ void k_pooldiv(float* __restrict__ outp, const float* __restrict__ cnt) {
  int i = blockIdx.x * blockDim.x + threadIdx.x;
  if (i >= NGRAPH * HD_) return;
  outp[i] = outp[i] / fmaxf(cnt[i >> 8], 1.0f);
}

// ---------------------------------------------------------------- launch
static inline unsigned blks(long n) { return (unsigned)((n + 255) / 256); }

extern "C" void kernel_launch(void* const* d_in, const int* in_sizes, int n_in,
                              void* d_out, int out_size, void* d_ws, size_t ws_size,
                              hipStream_t stream) {
  const float* x      = (const float*)d_in[0];
  const int*   ei     = (const int*)d_in[1];
  const float* eattr  = (const float*)d_in[2];
  const int*   batch  = (const int*)d_in[3];
  const float* Wn     = (const float*)d_in[4];
  const float* bn_b   = (const float*)d_in[5];
  const float* We     = (const float*)d_in[6];
  const float* be_b   = (const float*)d_in[7];
  const float* lin_w  = (const float*)d_in[8];
  const float* att_s  = (const float*)d_in[9];
  const float* att_d  = (const float*)d_in[10];
  const float* lin_ew = (const float*)d_in[11];
  const float* att_e  = (const float*)d_in[12];
  const float* cbias  = (const float*)d_in[13];
  const float* gamma  = (const float*)d_in[14];
  const float* beta   = (const float*)d_in[15];
  float* outp = (float*)d_out;

  char* w = (char*)d_ws;
  auto alloc = [&](size_t bytes) -> void* {
    void* p = (void*)w; w += (bytes + 255) & ~size_t(255); return p;
  };
  float*          hA    = (float*)alloc((size_t)NNODES * HD_ * 4);
  float*          hB    = (float*)alloc((size_t)NNODES * HD_ * 4);
  unsigned short* hbf   = (unsigned short*)alloc((size_t)NNODES * HD_ * 2);
  unsigned short* eabf  = (unsigned short*)alloc((size_t)ENTOT * HD_ * 2);
  float*          agg   = (float*)alloc((size_t)NNODES * HD_ * 4);   // also loop_sum
  float*          deg   = (float*)alloc((size_t)NNODES * 4);
  float*          xh    = (float*)alloc((size_t)NNODES * HD_ * 4);
  float*          aeC   = (float*)alloc((size_t)ENTOT * 16 * 4);
  float*          alpha = (float*)alloc((size_t)ENTOT * NHEAD * 4);
  float*          a_src = (float*)alloc((size_t)NNODES * NHEAD * 4);
  float*          a_dst = (float*)alloc((size_t)NNODES * NHEAD * 4);
  float*          seg_m = (float*)alloc((size_t)NNODES * NHEAD * 4);
  float*          seg_s = (float*)alloc((size_t)NNODES * NHEAD * 4);
  unsigned short* lwbf  = (unsigned short*)alloc((size_t)HD_ * HD_ * 2);
  unsigned short* webf  = (unsigned short*)alloc((size_t)HD_ * 16 * 2);
  float*          cnt   = (float*)alloc((size_t)NGRAPH * 4);

  // ---- embeddings + self-loop mean edge attr
  hipMemsetAsync(deg, 0, (size_t)NNODES * 4, stream);
  hipMemsetAsync(agg, 0, (size_t)NNODES * HD_ * 4, stream);
  k_node_embed<<<blks((long)NNODES * HD_), 256, 0, stream>>>(x, Wn, bn_b, hA);
  k_deg<<<blks(NEDGES), 256, 0, stream>>>(ei, deg);
  k_edge_embed<<<blks((long)NEDGES * HD_), 256, 0, stream>>>(eattr, We, be_b, ei, eabf, agg);
  k_loop_attr<<<blks((long)NNODES * HD_), 256, 0, stream>>>(agg, deg, eabf);

  float* h = hA; float* hn = hB;
  const unsigned gyN = (unsigned)((NNODES / 16 + 7) / 8);   // node GEMM wave-tiles
  const unsigned gyE = (unsigned)((ENTOT  / 16 + 7) / 8);   // edge-att GEMM wave-tiles

  for (int l = 0; l < NLAY; ++l) {
    k_f2bf<<<blks(HD_ * HD_), 256, 0, stream>>>(lin_w + (long)l * HD_ * HD_, lwbf, HD_ * HD_);
    k_f2bf<<<blks((long)NNODES * HD_), 256, 0, stream>>>(h, hbf, (long)NNODES * HD_);
    k_watt_e<<<blks(256 * 16), 256, 0, stream>>>(lin_ew + (long)l * HD_ * HD_,
                                                 att_e + l * NHEAD * 64, webf);
    // xh = h @ lin_w   (TDM-staged B, v_wmma_f32_16x16x32_bf16)
    k_gemm_bf16<HD_><<<dim3(16, gyN), 256, 0, stream>>>(hbf, lwbf, xh, NNODES);
    // a_e = ea @ w_att_e  (same WMMA kernel, 16-wide B, heads in cols 0..3)
    k_gemm_bf16<16><<<dim3(1, gyE), 256, 0, stream>>>(eabf, webf, aeC, ENTOT);

    k_att_nd<<<blks(NNODES * NHEAD), 256, 0, stream>>>(xh, att_s + l * NHEAD * 64,
                                                       att_d + l * NHEAD * 64, a_src, a_dst);
    k_fill<<<blks(NNODES * NHEAD), 256, 0, stream>>>(seg_m, -1e30f, NNODES * NHEAD);
    hipMemsetAsync(seg_s, 0, (size_t)NNODES * NHEAD * 4, stream);
    k_alpha<<<blks((long)ENTOT * NHEAD), 256, 0, stream>>>(ei, a_src, a_dst, aeC, alpha, seg_m);
    k_exp<<<blks((long)ENTOT * NHEAD), 256, 0, stream>>>(ei, alpha, seg_m, seg_s);
    k_wnorm<<<blks((long)ENTOT * NHEAD), 256, 0, stream>>>(ei, alpha, seg_s);

    hipMemsetAsync(agg, 0, (size_t)NNODES * HD_ * 4, stream);
    k_msg<<<blks((long)ENTOT * HD_), 256, 0, stream>>>(ei, xh, alpha, agg);
    k_bn_relu<<<blks((long)NNODES * HD_), 256, 0, stream>>>(agg, cbias + l * HD_,
                                                            gamma + l * HD_, beta + l * HD_, hn);
    float* t = h; h = hn; hn = t;
  }

  // ---- global mean pool
  hipMemsetAsync(d_out, 0, (size_t)NGRAPH * HD_ * 4, stream);
  hipMemsetAsync(cnt, 0, (size_t)NGRAPH * 4, stream);
  k_cnt<<<blks(NNODES), 256, 0, stream>>>(batch, cnt);
  k_pool<<<blks((long)NNODES * HD_), 256, 0, stream>>>(batch, h, outp);
  k_pooldiv<<<blks(NGRAPH * HD_), 256, 0, stream>>>(outp, cnt);
}
